// Normal_Conv_NLM_Shallow_59287728554347
// MI455X (gfx1250) — matrix operand order repaired and checked by reading, not compiled, not run on hardware
//
#include <hip/hip_runtime.h>

// ---------------------------------------------------------------------------
// CDNA5 (gfx1250) WMMA implementation of the Conv+NLM U-Net forward pass.
// All heavy math goes through v_wmma_f32_16x16x32_f16 (f32 accumulate).
// Activations live in NHWC f16; weights reordered to [oc][kh][kw][ic] f16.
// K loop unrolled x2 (Ktot is always a multiple of 64) with speculative
// global_prefetch_b8 for the next K-step's lines.
// ---------------------------------------------------------------------------

typedef __attribute__((ext_vector_type(16))) _Float16 v16h;
typedef __attribute__((ext_vector_type(8)))  float    v8f;

#define DEVINL __device__ __forceinline__

union V16U { uint4 u[2]; v16h v; };

DEVINL v16h frag_zero() {
  V16U c; c.u[0] = make_uint4(0u,0u,0u,0u); c.u[1] = make_uint4(0u,0u,0u,0u);
  return c.v;
}

// A-matrix (16x32 f16) fragment: lane hi-half selects K range
// {0..15} / {16..31}; halves [0..15] = K[2*phase8..2*phase8+15].
DEVINL v16h frag_load_A(const _Float16* __restrict__ base, int phase8) {
  V16U c;
  c.u[0] = *(const uint4*)(base + 2 * phase8);
  c.u[1] = *(const uint4*)(base + 2 * phase8 + 8);
  return c.v;
}

// B-matrix (32x16 f16) fragment: lane = column, hi-half selects K range
// {0..15} / {16..31}; 32 contiguous bytes per lane.
DEVINL v16h frag_load_B(const _Float16* __restrict__ base, int phase16) {
  const uint4* p = (const uint4*)(base + phase16);
  V16U c; c.u[0] = p[0]; c.u[1] = p[1];
  return c.v;
}

DEVINL v8f acc_zero() {
  v8f a;
#pragma unroll
  for (int i = 0; i < 8; ++i) a[i] = 0.0f;
  return a;
}

DEVINL v8f wmma16(v16h a, v16h b, v8f c) {
  return __builtin_amdgcn_wmma_f32_16x16x32_f16(false, a, false, b,
                                                (short)0, c, false, false);
}

// ---------------------------------------------------------------------------
// Implicit-GEMM convolution. act: [B*H*W][Cin] NHWC f16.
// wgt: [OC][KH*KW*Cin] f16 with K ordered (kh,kw,ic).  raw out: [Npix][OC] f32.
// Wave tile: M=32 pixels x N=64 out-channels; K unrolled x2 (16 WMMA / trip).
// ---------------------------------------------------------------------------
__global__ void __launch_bounds__(128)
conv_wmma(const _Float16* __restrict__ act, const _Float16* __restrict__ wgt,
          float* __restrict__ raw,
          int Npix, int H, int W, int Cin, int OC,
          int KH, int KW, int stride, int pad, int OH, int OW)
{
  int lane = threadIdx.x & 31;
  int wave = threadIdx.x >> 5;
  int m0 = (blockIdx.x * 4 + wave) * 32;
  if (m0 >= Npix) return;                       // wave-uniform: EXEC stays full
  int n0 = blockIdx.y * 64;
  int laneM = lane & 15;
  int hi = lane >> 4;
  int phase8 = hi * 8, phase16 = hi * 16;
  int OHW = OH * OW;

  int nIdx[2], ihb[2], iwb[2];
#pragma unroll
  for (int s = 0; s < 2; ++s) {
    int row = m0 + s * 16 + laneM;
    if (row >= Npix) row = Npix - 1;            // clamp; stores masked later
    int n = row / OHW; int rem = row - n * OHW;
    int oh = rem / OW; int ow = rem - oh * OW;
    nIdx[s] = n; ihb[s] = oh * stride - pad; iwb[s] = ow * stride - pad;
  }

  int Ktot = KH * KW * Cin;
  const _Float16* wptr[4];
#pragma unroll
  for (int t = 0; t < 4; ++t)
    wptr[t] = wgt + (size_t)(n0 + t * 16 + laneM) * Ktot + phase16;

  v8f acc[2][4];
#pragma unroll
  for (int s = 0; s < 2; ++s)
#pragma unroll
    for (int t = 0; t < 4; ++t) acc[s][t] = acc_zero();

  int kh = 0, kw = 0, ic0 = 0;                  // Cin and Ktot are multiples of 64
  for (int k0 = 0; k0 < Ktot; k0 += 64) {
    v16h a0[2], a1[2];
#pragma unroll
    for (int s = 0; s < 2; ++s) {
      int ih = ihb[s] + kh, iw = iwb[s] + kw;
      if (ih >= 0 && ih < H && iw >= 0 && iw < W) {
        const _Float16* p = act + (((size_t)nIdx[s] * H + ih) * W + iw) * Cin + ic0;
        a0[s] = frag_load_A(p, phase8);
        a1[s] = frag_load_A(p + 32, phase8);
        __builtin_prefetch(p + 64, 0, 1);       // speculative: next chunk / next row
      } else {
        a0[s] = frag_zero();
        a1[s] = frag_zero();
      }
    }
#pragma unroll
    for (int t = 0; t < 4; ++t) {
      v16h b0 = frag_load_B(wptr[t] + k0, 0);
      v16h b1 = frag_load_B(wptr[t] + k0 + 32, 0);
      __builtin_prefetch(wptr[t] + k0 + 64, 0, 1);
      acc[0][t] = wmma16(a0[0], b0, acc[0][t]);
      acc[1][t] = wmma16(a0[1], b0, acc[1][t]);
      acc[0][t] = wmma16(a1[0], b1, acc[0][t]);
      acc[1][t] = wmma16(a1[1], b1, acc[1][t]);
    }
    ic0 += 64;
    if (ic0 == Cin) { ic0 = 0; ++kw; if (kw == KW) { kw = 0; ++kh; } }
  }

#pragma unroll
  for (int s = 0; s < 2; ++s) {
    int rbase = m0 + s * 16 + hi * 8;           // C/D layout: row = r + 8*hi
#pragma unroll
    for (int t = 0; t < 4; ++t) {
      int col = n0 + t * 16 + laneM;
#pragma unroll
      for (int r = 0; r < 8; ++r) {
        int row = rbase + r;
        if (row < Npix) raw[(size_t)row * OC + col] = acc[s][t][r];
      }
    }
  }
}

// ---------------------------------------------------------------------------
// Generic NT GEMM (D = X * Y^T), f16 in / f32 accumulate, with fused NLM
// epilogues.  mode 1: Gram -> NLM weights w = exp(-sqrt(d2)/25)*bkg[col],
//                     stored f16 (padded cols zeroed) + row-sum atomics.
// mode 2: avg = acc/wsum[row]; out = bkg[row]>0 ? f[row][col] : avg  (f16).
// ---------------------------------------------------------------------------
__global__ void __launch_bounds__(128)
gemm_nt_wmma(const _Float16* __restrict__ X, int ldx,
             const _Float16* __restrict__ Y, int ldy,
             int M, int N, int ksteps, int mode,
             _Float16* __restrict__ w16, int ldw,
             const float* __restrict__ sq, const float* __restrict__ bkg,
             float* __restrict__ wsum,
             const _Float16* __restrict__ forig,
             _Float16* __restrict__ out16, int ldo)
{
  int lane = threadIdx.x & 31;
  int wave = threadIdx.x >> 5;
  int m0 = (blockIdx.x * 4 + wave) * 32;
  if (m0 >= M) return;
  int n0 = blockIdx.y * 64;
  int laneM = lane & 15;
  int hi = lane >> 4;
  int phase8 = hi * 8, phase16 = hi * 16;

  const _Float16* xptr[2];
#pragma unroll
  for (int s = 0; s < 2; ++s) {
    int row = m0 + s * 16 + laneM;
    if (row >= M) row = M - 1;
    xptr[s] = X + (size_t)row * ldx + phase8;
  }
  const _Float16* yptr[4];
#pragma unroll
  for (int t = 0; t < 4; ++t) {
    int col = n0 + t * 16 + laneM;
    if (col >= N) col = N - 1;
    yptr[t] = Y + (size_t)col * ldy + phase16;
  }

  v8f acc[2][4];
#pragma unroll
  for (int s = 0; s < 2; ++s)
#pragma unroll
    for (int t = 0; t < 4; ++t) acc[s][t] = acc_zero();

  for (int ks = 0; ks < ksteps; ++ks) {
    int k0 = ks * 32;
    v16h a[2];
#pragma unroll
    for (int s = 0; s < 2; ++s) {
      a[s] = frag_load_A(xptr[s] + k0 - phase8, phase8);
      __builtin_prefetch(xptr[s] + k0 + 32, 0, 1);
    }
#pragma unroll
    for (int t = 0; t < 4; ++t) {
      v16h b = frag_load_B(yptr[t] + k0, 0);
      __builtin_prefetch(yptr[t] + k0 + 32, 0, 1);
      acc[0][t] = wmma16(a[0], b, acc[0][t]);
      acc[1][t] = wmma16(a[1], b, acc[1][t]);
    }
  }

  if (mode == 1) {
    float sqc[4], bk[4]; int colv[4];
#pragma unroll
    for (int t = 0; t < 4; ++t) {
      int col = n0 + t * 16 + laneM;
      colv[t] = col;
      bool okc = col < N;
      sqc[t] = okc ? sq[col] : 0.0f;
      bk[t]  = okc ? bkg[col] : 0.0f;
    }
#pragma unroll
    for (int s = 0; s < 2; ++s) {
      int rbase = m0 + s * 16 + hi * 8;
#pragma unroll
      for (int r = 0; r < 8; ++r) {
        int row = rbase + r;
        bool okr = row < M;
        float sqr_ = okr ? sq[row] : 0.0f;
        float rs = 0.0f;
#pragma unroll
        for (int t = 0; t < 4; ++t) {
          float g  = acc[s][t][r];
          float d2 = sqr_ + sqc[t] - 2.0f * g;
          float d  = sqrtf(fmaxf(d2, 0.0f));
          float wv = expf(-d * (1.0f / 25.0f)) * bk[t];   // H_GAUSSIAN^2 = 25
          if (colv[t] >= N) wv = 0.0f;
          if (okr && colv[t] < ldw)
            w16[(size_t)row * ldw + colv[t]] = (_Float16)wv;
          rs += wv;
        }
        // reduce row-sum across the 16 lanes sharing this row
        for (int o = 1; o < 16; o <<= 1) rs += __shfl_xor(rs, o, 16);
        if (laneM == 0 && okr) atomicAdd(&wsum[row], rs);
      }
    }
  } else { // mode 2
#pragma unroll
    for (int s = 0; s < 2; ++s) {
      int rbase = m0 + s * 16 + hi * 8;
#pragma unroll
      for (int r = 0; r < 8; ++r) {
        int row = rbase + r;
        if (row >= M) continue;
        float denom = wsum[row];
        float bg = bkg[row];
#pragma unroll
        for (int t = 0; t < 4; ++t) {
          int col = n0 + t * 16 + laneM;
          if (col >= N) continue;
          float avg = acc[s][t][r] / denom;
          float fv  = (float)forig[(size_t)row * ldo + col];
          out16[(size_t)row * ldo + col] = (_Float16)(bg > 0.0f ? fv : avg);
        }
      }
    }
  }
}

// ---------------------------------------------------------------------------
// Elementwise / reduction helpers
// ---------------------------------------------------------------------------
__global__ void zero32(unsigned* __restrict__ p, size_t n) {
  size_t i = (size_t)blockIdx.x * blockDim.x + threadIdx.x;
  if (i < n) p[i] = 0u;
}

// NCHW f32 -> NHWC f16
__global__ void nchw2nhwc(const float* __restrict__ x, _Float16* __restrict__ y,
                          int Bn, int C, int H, int W) {
  int i = blockIdx.x * 256 + threadIdx.x;
  int tot = Bn * C * H * W;
  if (i >= tot) return;
  int w = i % W; int t = i / W;
  int h = t % H; t /= H;
  int c = t % C; int n = t / C;
  y[((size_t)(n * H + h) * W + w) * C + c] = (_Float16)x[i];
}

// weight [oc][ic][kh][kw] f32 -> [oc][(kh*KW+kw)*Cin+ic] f16
__global__ void wreorder(const float* __restrict__ w, _Float16* __restrict__ wt,
                         int OC, int Cin, int KH, int KW) {
  int i = blockIdx.x * 256 + threadIdx.x;
  int tot = OC * Cin * KH * KW;
  if (i >= tot) return;
  int kw = i % KW; int t = i / KW;
  int kh = t % KH; t /= KH;
  int ic = t % Cin; int oc = t / Cin;
  wt[(size_t)oc * (KH * KW * Cin) + (kh * KW + kw) * Cin + ic] = (_Float16)w[i];
}

// BN batch statistics over NHWC rows: per-channel sum / sumsq (atomics).
__global__ void __launch_bounds__(256)
bn_stats(const float* __restrict__ raw, float* __restrict__ bsum,
         float* __restrict__ bsq, int Npix, int OC) {
  __shared__ float s1[256], s2[256];
  int c = blockIdx.x * 64 + (threadIdx.x & 63);
  int rstart = (int)blockIdx.y * 4096;
  int rend = rstart + 4096; if (rend > Npix) rend = Npix;
  float a = 0.0f, b = 0.0f;
  for (int r = rstart + (threadIdx.x >> 6); r < rend; r += 4) {
    float v = raw[(size_t)r * OC + c];
    a += v; b += v * v;
  }
  s1[threadIdx.x] = a; s2[threadIdx.x] = b;
  __syncthreads();
  if (threadIdx.x < 64) {
    a = s1[threadIdx.x] + s1[threadIdx.x + 64] + s1[threadIdx.x + 128] + s1[threadIdx.x + 192];
    b = s2[threadIdx.x] + s2[threadIdx.x + 64] + s2[threadIdx.x + 128] + s2[threadIdx.x + 192];
    atomicAdd(&bsum[c], a);
    atomicAdd(&bsq[c], b);
  }
}

// Fused BN-apply / bias / activation / f32->f16.  act: 0 none, 1 relu, 2 lrelu
__global__ void pointwise(const float* __restrict__ raw, _Float16* __restrict__ out,
                          int total, int OC,
                          const float* __restrict__ bsum, const float* __restrict__ bsq,
                          float invN,
                          const float* __restrict__ gamma, const float* __restrict__ beta,
                          const float* __restrict__ bias, int act) {
  int i = blockIdx.x * 256 + threadIdx.x;
  if (i >= total) return;
  int c = i % OC;
  float x = raw[i];
  if (bsum) {
    float m = bsum[c] * invN;
    float v = bsq[c] * invN - m * m;
    x = gamma[c] * (x - m) * rsqrtf(v + 1e-5f) + beta[c];
  }
  if (bias) x += bias[c];
  if (act == 1) x = fmaxf(x, 0.0f);
  else if (act == 2) x = x > 0.0f ? x : 0.2f * x;
  out[i] = (_Float16)x;
}

// Final layer: raw NHWC f32 + bias -> d_out NCHW f32 (B=4,C=128,H=W=144)
__global__ void final_nchw(const float* __restrict__ raw,
                           const float* __restrict__ bias,
                           float* __restrict__ out) {
  int i = blockIdx.x * 256 + threadIdx.x;
  int tot = 4 * 128 * 144 * 144;
  if (i >= tot) return;
  int w = i % 144; int t = i / 144;
  int h = t % 144; t /= 144;
  int c = t % 128; int n = t / 128;
  out[i] = raw[((size_t)(n * 144 + h) * 144 + w) * 128 + c] + bias[c];
}

// nearest 2x upsample into channel slice [0,Cin) of concat dst
__global__ void up2_cat(const _Float16* __restrict__ in, _Float16* __restrict__ dst,
                        int Bn, int H, int W, int Cin, int Cdst) {
  int i = blockIdx.x * 256 + threadIdx.x;
  int H2 = 2 * H, W2 = 2 * W;
  int tot = Bn * H2 * W2 * Cin;
  if (i >= tot) return;
  int c = i % Cin; int p = i / Cin;
  int ow = p % W2; p /= W2;
  int oh = p % H2; int n = p / H2;
  dst[((size_t)(n * H2 + oh) * W2 + ow) * Cdst + c] =
      in[((size_t)(n * H + (oh >> 1)) * W + (ow >> 1)) * Cin + c];
}

// copy src [Npix][Cs] into dst [Npix][Cd] at channel offset coff
__global__ void cat_copy(const _Float16* __restrict__ src, _Float16* __restrict__ dst,
                         int Npix, int Cs, int Cd, int coff) {
  int i = blockIdx.x * 256 + threadIdx.x;
  int tot = Npix * Cs;
  if (i >= tot) return;
  int c = i % Cs; int p = i / Cs;
  dst[(size_t)p * Cd + coff + c] = src[i];
}

// get_feature_map stage 1: mask 144x144 -> 72x72 (7x7/49, stride2, pad3 ones)
__global__ void box_mask1(const float* __restrict__ mask, float* __restrict__ m1) {
  int i = blockIdx.x * 256 + threadIdx.x;
  if (i >= 4 * 72 * 72) return;
  int ow = i % 72; int t = i / 72;
  int oh = t % 72; int n = t / 72;
  float s = 0.0f;
  for (int kh = 0; kh < 7; ++kh)
    for (int kw = 0; kw < 7; ++kw) {
      int ih = oh * 2 - 3 + kh, iw = ow * 2 - 3 + kw;
      float v = (ih >= 0 && ih < 144 && iw >= 0 && iw < 144)
                    ? mask[((size_t)n * 144 + ih) * 144 + iw] : 1.0f;
      s += v;
    }
  m1[i] = s * (1.0f / 49.0f);
}

// stage 2: 72x72 -> 36x36 (5x5/25, stride2, pad2 ones) + threshold
__global__ void box_mask2(const float* __restrict__ m1, float* __restrict__ bkg) {
  int i = blockIdx.x * 256 + threadIdx.x;
  if (i >= 4 * 36 * 36) return;
  int ow = i % 36; int t = i / 36;
  int oh = t % 36; int n = t / 36;
  float s = 0.0f;
  for (int kh = 0; kh < 5; ++kh)
    for (int kw = 0; kw < 5; ++kw) {
      int ih = oh * 2 - 2 + kh, iw = ow * 2 - 2 + kw;
      float v = (ih >= 0 && ih < 72 && iw >= 0 && iw < 72)
                    ? m1[((size_t)n * 72 + ih) * 72 + iw] : 1.0f;
      s += v;
    }
  bkg[i] = (s * (1.0f / 25.0f) > (5.2f / 9.0f)) ? 1.0f : 0.0f;
}

// per-row sum of squares of f16 feature rows (one wave32 per row)
__global__ void rowsq(const _Float16* __restrict__ f, float* __restrict__ sq,
                      int rows, int C) {
  int gw = (blockIdx.x * blockDim.x + threadIdx.x) >> 5;
  int lane = threadIdx.x & 31;
  if (gw >= rows) return;
  const _Float16* p = f + (size_t)gw * C;
  float s = 0.0f;
  for (int c = lane; c < C; c += 32) {
    float v = (float)p[c];
    s += v * v;
  }
  for (int o = 16; o; o >>= 1) s += __shfl_xor(s, o, 32);
  if (lane == 0) sq[gw] = s;
}

// LDS-tiled f16 transpose: in [R][Cc] -> out [Cc][ldo] (pad pre-zeroed)
__global__ void __launch_bounds__(256)
transpose_h(const _Float16* __restrict__ in, _Float16* __restrict__ out,
            int R, int Cc, int ldo) {
  __shared__ _Float16 tile[16][17];
  int tx = threadIdx.x & 15, ty = threadIdx.x >> 4;
  int r0 = blockIdx.y * 16, c0 = blockIdx.x * 16;
  int r = r0 + ty, c = c0 + tx;
  tile[ty][tx] = (r < R && c < Cc) ? in[(size_t)r * Cc + c] : (_Float16)0.0f;
  __syncthreads();
  int orow = c0 + ty, ocol = r0 + tx;
  if (orow < Cc && ocol < R) out[(size_t)orow * ldo + ocol] = tile[tx][ty];
}

// ---------------------------------------------------------------------------
// Host orchestration
// ---------------------------------------------------------------------------
static inline int cdiv(int a, int b) { return (a + b - 1) / b; }

extern "C" void kernel_launch(void* const* d_in, const int* in_sizes, int n_in,
                              void* d_out, int out_size, void* d_ws, size_t ws_size,
                              hipStream_t stream) {
  (void)in_sizes; (void)n_in; (void)out_size; (void)ws_size;
  const float* x_in  = (const float*)d_in[0];   // (4,128,144,144)
  const float* maskp = (const float*)d_in[1];   // (4,1,144,144)

  const int B = 4;
  char* wsb = (char*)d_ws;
  size_t off = 0;
  auto carve = [&](size_t bytes) -> void* {
    void* p = wsb + off;
    off = (off + bytes + 255) & ~(size_t)255;
    return p;
  };

  // f16 activation buffers (NHWC)
  _Float16* x16   = (_Float16*)carve((size_t)B * 144 * 144 * 128 * 2);
  _Float16* pre0  = (_Float16*)carve((size_t)B * 144 * 144 * 128 * 2);
  _Float16* pre1  = (_Float16*)carve((size_t)B * 144 * 144 * 128 * 2);
  _Float16* b1    = (_Float16*)carve((size_t)B * 72 * 72 * 256 * 2);
  _Float16* feat  = (_Float16*)carve((size_t)B * 36 * 36 * 512 * 2);   // enc2 act (NLM f)
  _Float16* buf2  = (_Float16*)carve((size_t)B * 36 * 36 * 512 * 2);   // NLM out
  _Float16* b3    = (_Float16*)carve((size_t)B * 18 * 18 * 512 * 2);
  _Float16* e4a   = (_Float16*)carve((size_t)B * 9 * 9 * 512 * 2);
  _Float16* S1    = (_Float16*)carve((size_t)B * 144 * 144 * 384 * 2); // concat scratch
  _Float16* S2    = (_Float16*)carve((size_t)B * 144 * 144 * 128 * 2); // dec-out scratch
  float*    raw   = (float*)carve((size_t)B * 144 * 144 * 128 * 4);    // conv f32 out
  // NLM buffers
  const int Nn = 1296, Kpad = 1312, Cf = 512;
  _Float16* w16 = (_Float16*)carve((size_t)B * Nn * Kpad * 2);
  _Float16* fT  = (_Float16*)carve((size_t)B * Cf * Kpad * 2);
  float* sq   = (float*)carve((size_t)B * Nn * 4);
  float* wsum = (float*)carve((size_t)B * Nn * 4);
  float* bkg  = (float*)carve((size_t)B * Nn * 4);
  float* m1   = (float*)carve((size_t)B * 72 * 72 * 4);
  float* bnsum = (float*)carve(1024 * 4);       // bnsum[512] + bnsq[512]
  float* bnsq = bnsum + 512;

  // f16 weights
  struct WCfg { int idx, OC, Cin, KH, KW; _Float16* wt; };
  WCfg wc[12] = {
    { 2, 128, 128, 3, 3, nullptr},  // enc_pre0
    { 3, 128, 128, 3, 3, nullptr},  // enc_pre1
    { 4, 256, 128, 7, 7, nullptr},  // enc1
    { 5, 512, 256, 5, 5, nullptr},  // enc2
    { 8, 512, 512, 3, 3, nullptr},  // enc3
    {11, 512, 512, 3, 3, nullptr},  // enc4
    {14, 512,1024, 3, 3, nullptr},  // dec4
    {17, 512,1024, 3, 3, nullptr},  // dec3
    {20, 256, 768, 3, 3, nullptr},  // dec2
    {23, 128, 384, 3, 3, nullptr},  // dec1
    {25, 128, 256, 3, 3, nullptr},  // decp1
    {27, 128, 256, 3, 3, nullptr},  // decp0
  };
  for (int i = 0; i < 12; ++i) {
    size_t e = (size_t)wc[i].OC * wc[i].Cin * wc[i].KH * wc[i].KW;
    wc[i].wt = (_Float16*)carve(e * 2);
  }

  auto zero = [&](void* p, size_t bytes) {
    size_t nw = bytes / 4;
    zero32<<<dim3((unsigned)((nw + 255) / 256)), 256, 0, stream>>>((unsigned*)p, nw);
  };
  auto conv = [&](const _Float16* in, const _Float16* wt, int H, int W, int Cin,
                  int OC, int KH, int KW, int st, int pd, int OH, int OW) {
    int Npix = B * OH * OW;
    dim3 g(cdiv(cdiv(Npix, 32), 4), OC / 64);
    conv_wmma<<<g, 128, 0, stream>>>(in, wt, raw, Npix, H, W, Cin, OC,
                                     KH, KW, st, pd, OH, OW);
  };
  auto stats = [&](int Npix, int OC) {
    zero(bnsum, 1024 * 4);
    dim3 g(OC / 64, cdiv(Npix, 4096));
    bn_stats<<<g, 256, 0, stream>>>(raw, bnsum, bnsq, Npix, OC);
  };
  auto pw = [&](_Float16* o16, int Npix, int OC, bool bn,
                const float* gam, const float* bet, const float* bias, int act) {
    int tot = Npix * OC;
    pointwise<<<cdiv(tot, 256), 256, 0, stream>>>(
        raw, o16, tot, OC, bn ? bnsum : nullptr, bn ? bnsq : nullptr,
        bn ? 1.0f / (float)Npix : 0.0f, gam, bet, bias, act);
  };
  auto catc = [&](const _Float16* src, _Float16* dst, int Npix, int Cs, int Cd, int coff) {
    int tot = Npix * Cs;
    cat_copy<<<cdiv(tot, 256), 256, 0, stream>>>(src, dst, Npix, Cs, Cd, coff);
  };
  auto up2 = [&](const _Float16* in, _Float16* dst, int H, int W, int Cin, int Cd) {
    int tot = B * 4 * H * W * Cin;
    up2_cat<<<cdiv(tot, 256), 256, 0, stream>>>(in, dst, B, H, W, Cin, Cd);
  };

  // -- weight reorder + input convert ---------------------------------------
  for (int i = 0; i < 12; ++i) {
    int tot = wc[i].OC * wc[i].Cin * wc[i].KH * wc[i].KW;
    wreorder<<<cdiv(tot, 256), 256, 0, stream>>>((const float*)d_in[wc[i].idx],
                                                 wc[i].wt, wc[i].OC, wc[i].Cin,
                                                 wc[i].KH, wc[i].KW);
  }
  {
    int tot = B * 128 * 144 * 144;
    nchw2nhwc<<<cdiv(tot, 256), 256, 0, stream>>>(x_in, x16, B, 128, 144, 144);
  }

  // -- encoder --------------------------------------------------------------
  conv(x16,  wc[0].wt, 144, 144, 128, 128, 3, 3, 1, 1, 144, 144);
  pw(pre0, B * 144 * 144, 128, false, nullptr, nullptr, nullptr, 1);
  conv(pre0, wc[1].wt, 144, 144, 128, 128, 3, 3, 1, 1, 144, 144);
  pw(pre1, B * 144 * 144, 128, false, nullptr, nullptr, nullptr, 1);
  conv(pre1, wc[2].wt, 144, 144, 128, 256, 7, 7, 2, 3, 72, 72);
  pw(b1, B * 72 * 72, 256, false, nullptr, nullptr, nullptr, 1);
  conv(b1,   wc[3].wt, 72, 72, 256, 512, 5, 5, 2, 2, 36, 36);
  stats(B * 36 * 36, 512);
  pw(feat, B * 36 * 36, 512, true, (const float*)d_in[6], (const float*)d_in[7], nullptr, 1);

  // -- get_feature_map ------------------------------------------------------
  box_mask1<<<cdiv(B * 72 * 72, 256), 256, 0, stream>>>(maskp, m1);
  box_mask2<<<cdiv(B * 36 * 36, 256), 256, 0, stream>>>(m1, bkg);

  // -- batch NLM (two WMMA GEMMs per batch, fused epilogues) ----------------
  rowsq<<<cdiv(B * Nn * 32, 256), 256, 0, stream>>>(feat, sq, B * Nn, Cf);
  zero(wsum, (size_t)B * Nn * 4);
  zero(fT, (size_t)B * Cf * Kpad * 2);
  for (int b = 0; b < B; ++b) {
    const _Float16* fb = feat + (size_t)b * Nn * Cf;
    _Float16* wb = w16 + (size_t)b * Nn * Kpad;
    _Float16* fTb = fT + (size_t)b * Cf * Kpad;
    // Gram + weight epilogue: D[n][m] = f.f^T -> w16 + wsum
    dim3 g1(cdiv(cdiv(Nn, 32), 4), cdiv(Nn, 64));
    gemm_nt_wmma<<<g1, 128, 0, stream>>>(fb, Cf, fb, Cf, Nn, Nn, Cf / 32, 1,
                                         wb, Kpad, sq + b * Nn, bkg + b * Nn,
                                         wsum + b * Nn, nullptr, nullptr, 0);
    // f^T for the NN-shaped second GEMM
    dim3 gt(Cf / 16, Nn / 16);
    transpose_h<<<gt, 256, 0, stream>>>(fb, fTb, Nn, Cf, Kpad);
    // avg = (w . f) / wsum; blend with f by bkg
    dim3 g2(cdiv(cdiv(Nn, 32), 4), Cf / 64);
    gemm_nt_wmma<<<g2, 128, 0, stream>>>(wb, Kpad, fTb, Kpad, Nn, Cf, Kpad / 32, 2,
                                         nullptr, 0, nullptr, bkg + b * Nn,
                                         wsum + b * Nn, fb,
                                         buf2 + (size_t)b * Nn * Cf, Cf);
  }

  // -- rest of encoder ------------------------------------------------------
  conv(buf2, wc[4].wt, 36, 36, 512, 512, 3, 3, 2, 1, 18, 18);
  stats(B * 18 * 18, 512);
  pw(b3, B * 18 * 18, 512, true, (const float*)d_in[9], (const float*)d_in[10], nullptr, 1);
  conv(b3, wc[5].wt, 18, 18, 512, 512, 3, 3, 2, 1, 9, 9);
  stats(B * 9 * 9, 512);
  pw(e4a, B * 9 * 9, 512, true, (const float*)d_in[12], (const float*)d_in[13], nullptr, 1);

  // -- decoder --------------------------------------------------------------
  up2(e4a, S1, 9, 9, 512, 1024);
  catc(b3, S1, B * 18 * 18, 512, 1024, 512);
  conv(S1, wc[6].wt, 18, 18, 1024, 512, 3, 3, 1, 1, 18, 18);
  stats(B * 18 * 18, 512);
  pw(S2, B * 18 * 18, 512, true, (const float*)d_in[15], (const float*)d_in[16], nullptr, 2);

  up2(S2, S1, 18, 18, 512, 1024);
  catc(buf2, S1, B * 36 * 36, 512, 1024, 512);
  conv(S1, wc[7].wt, 36, 36, 1024, 512, 3, 3, 1, 1, 36, 36);
  stats(B * 36 * 36, 512);
  pw(S2, B * 36 * 36, 512, true, (const float*)d_in[18], (const float*)d_in[19], nullptr, 2);

  up2(S2, S1, 36, 36, 512, 768);
  catc(b1, S1, B * 72 * 72, 256, 768, 512);
  conv(S1, wc[8].wt, 72, 72, 768, 256, 3, 3, 1, 1, 72, 72);
  stats(B * 72 * 72, 256);
  pw(S2, B * 72 * 72, 256, true, (const float*)d_in[21], (const float*)d_in[22], nullptr, 2);

  up2(S2, S1, 72, 72, 256, 384);
  catc(pre1, S1, B * 144 * 144, 128, 384, 256);
  conv(S1, wc[9].wt, 144, 144, 384, 128, 3, 3, 1, 1, 144, 144);
  pw(S2, B * 144 * 144, 128, false, nullptr, nullptr, (const float*)d_in[24], 2);

  catc(S2, S1, B * 144 * 144, 128, 256, 0);
  catc(pre0, S1, B * 144 * 144, 128, 256, 128);
  conv(S1, wc[10].wt, 144, 144, 256, 128, 3, 3, 1, 1, 144, 144);
  stats(B * 144 * 144, 128);
  pw(S2, B * 144 * 144, 128, true, (const float*)d_in[26], (const float*)d_in[27], nullptr, 2);

  catc(S2, S1, B * 144 * 144, 128, 256, 0);
  catc(x16, S1, B * 144 * 144, 128, 256, 128);
  conv(S1, wc[11].wt, 144, 144, 256, 128, 3, 3, 1, 1, 144, 144);
  {
    int tot = B * 128 * 144 * 144;
    final_nchw<<<cdiv(tot, 256), 256, 0, stream>>>(raw, (const float*)d_in[29],
                                                   (float*)d_out);
  }
}
